// STEGNN_23295902614278
// MI455X (gfx1250) — compile-verified, matrix operand order checked by reading, not run on hardware
//
#include <hip/hip_runtime.h>
#include <hip/hip_bf16.h>
#include <stdint.h>

// ---- problem constants (match reference) ----
#define NPT   10      // num_past timesteps
#define NNODE 2000
#define NEDGE 20000
#define PPART 20
#define HDIM  64
#define NLAY  4
#define INNF  8
#define EFD   2
#define RALPHA 0.5f
#define K1PAD 160     // edge-MLP concat K: 131 padded to multiple of 32
#define KNPAD 128     // node-MLP concat K

typedef __attribute__((ext_vector_type(16))) _Float16 v16h;
typedef __attribute__((ext_vector_type(8)))  float    v8f;

// ---------------- device helpers ----------------
__device__ __forceinline__ float silu_f(float v) { return v / (1.0f + __expf(-v)); }
__device__ __forceinline__ float softplus_f(float v) {
    return (v > 20.0f) ? v : __logf(1.0f + __expf(v));
}
__device__ __forceinline__ uint32_t hash_u32(uint32_t x) {
    x ^= x >> 16; x *= 0x7feb352du; x ^= x >> 15; x *= 0x846ca68bu; x ^= x >> 16;
    return x;
}
__device__ __forceinline__ float rng_uniform(uint32_t a, uint32_t b, uint32_t c) {
    uint32_t s = hash_u32(a * 0x9E3779B9u ^ hash_u32(b * 0x85EBCA6Bu ^ (c + 0x1234567u)));
    return ((float)(s >> 8) + 0.5f) * (1.0f / 16777216.0f);
}
__device__ __forceinline__ float rng_normal(uint32_t a, uint32_t b, uint32_t c) {
    float u1 = rng_uniform(a, b, c);
    float u2 = rng_uniform(a ^ 0x5bd1e995u, b, c);
    return sqrtf(fmaxf(-2.0f * __logf(u1), 0.0f)) * __cosf(6.28318530718f * u2);
}

// Swizzled A/activation position: within each 32-wide K chunk, swap the two
// middle 8-groups so a lane's WMMA A-fragment (half selects K-subgroups
// {h*8..h*8+7, 16+h*8..16+h*8+7}) is 16 contiguous halfs at chunk + h*16.
__device__ __forceinline__ int posA(int j) {
    int jj = j & 31, g = jj & 24;
    int pj = (g == 8) ? jj + 8 : (g == 16) ? jj - 8 : jj;
    return (j & ~31) + pj;
}

// Wave-level 16xK @ Kx64 GEMM via v_wmma_f32_16x16x32_f16.
// Ah : LDS f16, swizzled layout, lane fragment = contiguous v16h.
// Wsw: global f16, blocked [Kpad/32][4 ntiles][32 lanes][16] -> one v16h per lane.
// outF: optional LDS fp32 [16][64]; outH: optional LDS f16 swizzled [16][64].
__device__ __forceinline__ void wave_gemm_h(const _Float16* __restrict__ Ah, int Kpad,
                                            const _Float16* __restrict__ Wsw,
                                            const float* __restrict__ bias, int do_silu,
                                            float* __restrict__ outF,
                                            _Float16* __restrict__ outH, int lane)
{
    const int half = lane >> 4;
    const int m    = lane & 15;
    for (int nt = 0; nt < 4; ++nt) {
        v8f c = {};
        for (int kc = 0; kc < Kpad; kc += 32) {
            v16h a = *(const v16h*)(Ah + m * Kpad + kc + half * 16);
            v16h b = *(const v16h*)(Wsw + ((kc >> 5) * 4 + nt) * 512 + lane * 16);
            c = __builtin_amdgcn_wmma_f32_16x16x32_f16(
                    false, a, false, b, (short)0, c, false, false);
        }
        const int  col = nt * 16 + m;
        const float bv = bias[col];
        const int  pcol = posA(col);
#pragma unroll
        for (int r = 0; r < 8; ++r) {
            float v = c[r] + bv;
            if (do_silu) v = silu_f(v);
            int row = r + 8 * half;
            if (outF) outF[row * 64 + col]  = v;
            if (outH) outH[row * 64 + pcol] = (_Float16)v;
        }
    }
}

// ---------------- small kernels ----------------
__global__ void zero_kernel(float* p, int n) {
    int i = blockIdx.x * blockDim.x + threadIdx.x;
    if (i < n) p[i] = 0.0f;
}
__global__ void copy_kernel(float* dst, const float* src, int n) {
    int i = blockIdx.x * blockDim.x + threadIdx.x;
    if (i < n) dst[i] = src[i];
}
// fp32 [K][64] -> f16 blocked-swizzled [Kpad/32][4][32][16] (zero-padded K rows)
__global__ void convert_swizzle_kernel(const float* __restrict__ src,
                                       _Float16* __restrict__ dst, int K, int Kpad) {
    int idx = blockIdx.x * blockDim.x + threadIdx.x;
    if (idx >= Kpad * 64) return;
    int blk = idx >> 9, within = idx & 511;
    int lane = within >> 4, t = within & 15;
    int h = lane >> 4, n = lane & 15;
    int c0 = blk >> 2, nt = blk & 3;
    int jj = h * 8 + (t & 7) + ((t & 8) ? 16 : 0);
    int k  = c0 * 32 + jj;
    int col = nt * 16 + n;
    dst[idx] = (k < K) ? (_Float16)src[k * 64 + col] : (_Float16)0.0f;
}
__global__ void count_kernel(const int* __restrict__ erow, float* __restrict__ cnt) {
    int e = blockIdx.x * blockDim.x + threadIdx.x;
    if (e < NEDGE) atomicAdd(&cnt[erow[e]], 1.0f);
}
__global__ void clip_cnt_kernel(float* cnt) {
    int n = blockIdx.x * blockDim.x + threadIdx.x;
    if (n < NNODE) cnt[n] = fmaxf(cnt[n], 1.0f);
}
__global__ void init_xp_kernel(const float* __restrict__ x, float* __restrict__ xp) {
    int idx = blockIdx.x * blockDim.x + threadIdx.x;             // [N][T][3]
    if (idx >= NNODE * NPT * 3) return;
    int n = idx / (NPT * 3), rem = idx - n * NPT * 3;
    int t = rem / 3, d = rem - t * 3;
    xp[idx] = x[(t * NNODE + n) * 3 + d];
}
__global__ void init_coord_lw_kernel(const float* __restrict__ x,
                                     float* __restrict__ coord, float* __restrict__ lw) {
    int q = blockIdx.x * blockDim.x + threadIdx.x;               // (n,p)
    if (q >= NNODE * PPART) return;
    int n = q / PPART;
    lw[q] = 1.0f / PPART;
    for (int d = 0; d < 3; ++d) coord[q * 3 + d] = x[n * 3 + d]; // x[0] slice
}
__global__ void embed_kernel(const float* __restrict__ hin, const float* __restrict__ emb_w,
                             const float* __restrict__ emb_b, const float* __restrict__ temb,
                             float* __restrict__ hh) {
    int idx = blockIdx.x * blockDim.x + threadIdx.x;             // [N][T][64]
    if (idx >= NNODE * NPT * 64) return;
    int n = idx / (NPT * 64), rem = idx - n * NPT * 64;
    int t = rem >> 6, j = rem & 63;
    float v = emb_b[j] + temb[t * 64 + j];
    for (int i = 0; i < INNF; ++i) v += hin[n * INNF + i] * emb_w[i * 64 + j];
    hh[idx] = v;
}
__global__ void bcast_kernel(float* __restrict__ hcur, const float* __restrict__ hh, int k) {
    int idx = blockIdx.x * blockDim.x + threadIdx.x;             // [(n,p)][64]
    if (idx >= NNODE * PPART * 64) return;
    int q = idx >> 6, j = idx & 63;
    int n = q / PPART;
    hcur[idx] = hh[(n * NPT + k) * 64 + j];
}

// ---- edge GCL: edge MLP + coord/var heads + atomic segment scatter ----
// per-wave arena: [0,5120) Ah f16 16xK1PAD (reused as head fp32 buf after gemm1)
//                 [5120,7168) Bh f16 16x64 ; [7168,9216) Mh f16 16x64
//                 [9216,13312) F fp32 16x64 (m)
#define EW_BYTES 13312
__global__ void __launch_bounds__(128)
edge_gcl_kernel(const float* __restrict__ h, const float* __restrict__ coord,
                const int* __restrict__ erow, const int* __restrict__ ecol,
                const float* __restrict__ eattr,
                const _Float16* We1, const float* be1,
                const _Float16* We2, const float* be2,
                const _Float16* Wc1, const float* bc1, const float* Wc2,
                const _Float16* Wv1, const float* bv1, const float* Wv2,
                float* __restrict__ cagg, float* __restrict__ vagg,
                float* __restrict__ nagg)
{
    __shared__ __align__(32) char smem[4][EW_BYTES];
    __shared__ float sCD[4][16][3];
    __shared__ float sRad[4][16];
    __shared__ int   sE[4][16], sRN[4][16], sCN[4][16], sPI[4][16], sV[4][16];
    __shared__ float sW[4][16], sWV[4][16];

    const int lane = threadIdx.x & 31;
    const int w    = threadIdx.x >> 5;
    const int base = (blockIdx.x * 4 + w) * 16;
    const int EP   = NEDGE * PPART;

    _Float16* pAh = (_Float16*)(smem[w]);
    _Float16* pBh = (_Float16*)(smem[w] + 5120);
    _Float16* pMh = (_Float16*)(smem[w] + 7168);
    float*    pF  = (float*)(smem[w] + 9216);
    float*    pG  = (float*)(smem[w]);            // aliases Ah (safe after gemm1)

    if (lane < 16) {
        int q = base + lane;
        int valid = (q < EP);
        if (!valid) q = EP - 1;
        int e = q / PPART, p = q - e * PPART;
        int rn = erow[e], cn = ecol[e];
        sE[w][lane] = e; sRN[w][lane] = rn; sCN[w][lane] = cn;
        sPI[w][lane] = p; sV[w][lane] = valid;
        float rad = 0.0f;
        for (int d = 0; d < 3; ++d) {
            float cd = coord[(rn * PPART + p) * 3 + d] - coord[(cn * PPART + p) * 3 + d];
            sCD[w][lane][d] = cd;
            rad += cd * cd;
        }
        sRad[w][lane] = rad;
    }
    __syncthreads();

    // stage concat [h_row | h_col | radial | eattr | 0-pad] as swizzled f16
    for (int idx = lane; idx < 16 * K1PAD; idx += 32) {
        int r = idx / K1PAD, j = idx - r * K1PAD;
        float v;
        if (j < 64)        v = h[(sRN[w][r] * PPART + sPI[w][r]) * 64 + j];
        else if (j < 128)  v = h[(sCN[w][r] * PPART + sPI[w][r]) * 64 + (j - 64)];
        else if (j == 128) v = sRad[w][r];
        else if (j < 131)  v = eattr[sE[w][r] * EFD + (j - 129)];
        else               v = 0.0f;
        pAh[r * K1PAD + posA(j)] = (_Float16)v;
    }
    __syncthreads();

    wave_gemm_h(pAh, K1PAD, We1, be1, 1, nullptr, pBh, lane);  // 131->64 silu
    __syncthreads();
    wave_gemm_h(pBh, 64, We2, be2, 1, pF, pMh, lane);          // m: fp32 + f16
    __syncthreads();

    wave_gemm_h(pMh, 64, Wc1, bc1, 1, pG, nullptr, lane);      // coord head hidden
    __syncthreads();
    if (lane < 16) {
        float acc = 0.0f;
        for (int j = 0; j < 64; ++j) acc += pG[lane * 64 + j] * Wc2[j];
        sW[w][lane] = acc;
    }
    __syncthreads();
    wave_gemm_h(pMh, 64, Wv1, bv1, 1, pG, nullptr, lane);      // var head hidden
    __syncthreads();
    if (lane < 16) {
        float acc = 0.0f;
        for (int j = 0; j < 64; ++j) acc += pG[lane * 64 + j] * Wv2[j];
        sWV[w][lane] = acc;
    }
    __syncthreads();

    // scatter segment sums (fp32 L2 atomics)
    for (int idx = lane; idx < 16 * 64; idx += 32) {
        int r = idx >> 6, j = idx & 63;
        if (sV[w][r])
            atomicAdd(&nagg[(sRN[w][r] * PPART + sPI[w][r]) * 64 + j], pF[r * 64 + j]);
    }
    if (lane < 16 && sV[w][lane]) {
        int o = (sRN[w][lane] * PPART + sPI[w][lane]) * 3;
        for (int d = 0; d < 3; ++d) {
            atomicAdd(&cagg[o + d], sCD[w][lane][d] * sW[w][lane]);
            atomicAdd(&vagg[o + d], sCD[w][lane][d] * sWV[w][lane]);
        }
    }
}

// ---- node GCL: node MLP residual + coord/var update + reparameterize + agg re-zero ----
// per-wave arena: [0,4096) Ah f16 16xKNPAD ; [4096,6144) Bh f16 16x64 ; [6144,10240) F fp32
#define NW_BYTES 10240
__global__ void __launch_bounds__(128)
node_gcl_kernel(float* __restrict__ h, float* __restrict__ coord,
                const float* __restrict__ cnt,
                float* __restrict__ cagg, float* __restrict__ vagg,
                float* __restrict__ nagg,
                const _Float16* Wn1, const float* bn1,
                const _Float16* Wn2, const float* bn2, int seed)
{
    __shared__ __align__(32) char smem[4][NW_BYTES];
    const int lane = threadIdx.x & 31;
    const int w    = threadIdx.x >> 5;
    const int base = (blockIdx.x * 4 + w) * 16;
    const int NPQ  = NNODE * PPART;

    _Float16* pAh = (_Float16*)(smem[w]);
    _Float16* pBh = (_Float16*)(smem[w] + 4096);
    float*    pF  = (float*)(smem[w] + 6144);

    for (int idx = lane; idx < 16 * KNPAD; idx += 32) {
        int r = idx >> 7, j = idx & 127;
        int q = base + r; if (q >= NPQ) q = NPQ - 1;
        float v = (j < 64) ? h[q * 64 + j] : nagg[q * 64 + (j - 64)];
        pAh[r * KNPAD + posA(j)] = (_Float16)v;
    }
    __syncthreads();
    wave_gemm_h(pAh, KNPAD, Wn1, bn1, 1, nullptr, pBh, lane);  // 128->64 silu
    __syncthreads();
    wave_gemm_h(pBh, 64, Wn2, bn2, 0, pF, nullptr, lane);      // dh (no act)
    __syncthreads();
    for (int idx = lane; idx < 16 * 64; idx += 32) {
        int r = idx >> 6, j = idx & 63;
        int q = base + r;
        if (q < NPQ) {
            h[q * 64 + j] += pF[r * 64 + j];
            nagg[q * 64 + j] = 0.0f;                    // re-zero for next layer/step
        }
    }
    if (lane < 16) {
        int q = base + lane;
        if (q < NPQ) {
            int n = q / PPART;
            float c = cnt[n];
            for (int d = 0; d < 3; ++d) {
                float var = vagg[q * 3 + d] / c;
                float eps = rng_normal((uint32_t)seed, (uint32_t)q, (uint32_t)d);
                coord[q * 3 + d] += cagg[q * 3 + d] / c + eps * softplus_f(var);
                cagg[q * 3 + d] = 0.0f;
                vagg[q * 3 + d] = 0.0f;
            }
        }
    }
}

// ---- weight update + normalize (per node) ----
__global__ void weight_kernel(const float* __restrict__ wu_w, const float* __restrict__ wu_b,
                              const float* __restrict__ coord, const float* __restrict__ xp,
                              const float* __restrict__ lw_in, float* __restrict__ lw_out,
                              int k)
{
    int n = blockIdx.x * blockDim.x + threadIdx.x;
    if (n >= NNODE) return;
    float wuw = wu_w[0], wub = wu_b[0];
    float v[PPART]; float mx = -1e30f;
    for (int p = 0; p < PPART; ++p) {
        float s = 0.0f;
        for (int d = 0; d < 3; ++d)
            s += coord[(n * PPART + p) * 3 + d] - xp[(n * NPT + k + 1) * 3 + d];
        float t = lw_in[n * PPART + p] + s * wuw + wub;
        t = fminf(fmaxf(t, 0.0f), 6.0f) - 3.0f;        // relu6 - 3
        v[p] = t; mx = fmaxf(mx, t);
    }
    float se = 0.0f;
    for (int p = 0; p < PPART; ++p) se += __expf(v[p] - mx);
    float lse = mx + __logf(se);
    for (int p = 0; p < PPART; ++p) lw_out[n * PPART + p] = v[p] - lse;
}

// ---- soft resampling (replicates reference flat-index quirk) + xp[:,k] ----
__global__ void sample_kernel(const float* __restrict__ coord, float* __restrict__ coord2,
                              const float* __restrict__ lw2, float* __restrict__ lw_out,
                              float* __restrict__ xp, int k)
{
    int n = blockIdx.x * blockDim.x + threadIdx.x;
    if (n >= NNODE) return;
    float probs[PPART]; float tot = 0.0f;
    for (int p = 0; p < PPART; ++p) {
        probs[p] = RALPHA * __expf(lw2[n * PPART + p]) + (1.0f - RALPHA) / PPART;
        tot += probs[p];
    }
    float pc[PPART][3]; float nlw[PPART]; float mx = -1e30f;
    for (int po = 0; po < PPART; ++po) {
        float u = rng_uniform(10000u + (uint32_t)k, (uint32_t)n, (uint32_t)po) * tot;
        int idx = 0; float cs = probs[0];
        while (idx < PPART - 1 && u > cs) { ++idx; cs += probs[idx]; }
        int g  = n + idx * NNODE;          // reference: offset + idx*batch
        int r0 = g / PPART, r1 = g - r0 * PPART;
        for (int d = 0; d < 3; ++d) pc[po][d] = coord[(r0 * PPART + r1) * 3 + d];
        float pr = __expf(lw2[r0 * PPART + r1]);
        pr = pr / (RALPHA * pr + (1.0f - RALPHA) / PPART);
        nlw[po] = __logf(pr); mx = fmaxf(mx, nlw[po]);
    }
    float se = 0.0f;
    for (int po = 0; po < PPART; ++po) se += __expf(nlw[po] - mx);
    float lse = mx + __logf(se);
    float xc[3] = {0.0f, 0.0f, 0.0f};
    for (int po = 0; po < PPART; ++po) {
        nlw[po] -= lse;
        lw_out[n * PPART + po] = nlw[po];
        float wgt = __expf(nlw[po]);
        for (int d = 0; d < 3; ++d) {
            coord2[(n * PPART + po) * 3 + d] = pc[po][d];
            xc[d] += pc[po][d] * wgt;
        }
    }
    for (int d = 0; d < 3; ++d) xp[(n * NPT + k) * 3 + d] = xc[d];
}

__global__ void final_kernel(const float* __restrict__ theta, const float* __restrict__ xp,
                             float* __restrict__ out)
{
    int idx = blockIdx.x * blockDim.x + threadIdx.x;
    if (idx >= NNODE * 3) return;
    int n = idx / 3, d = idx - 3 * n;
    float last = xp[(n * NPT + NPT - 1) * 3 + d];
    float acc = last;
    for (int t = 0; t < NPT; ++t)
        acc += theta[t] * (xp[(n * NPT + t) * 3 + d] - last);
    out[idx] = acc;
}

// ---------------- host launcher ----------------
extern "C" void kernel_launch(void* const* d_in, const int* in_sizes, int n_in,
                              void* d_out, int out_size, void* d_ws, size_t ws_size,
                              hipStream_t stream)
{
    (void)in_sizes; (void)n_in; (void)out_size; (void)ws_size;
    const float* h_in   = (const float*)d_in[0];
    const float* x_in   = (const float*)d_in[1];
    const int*   erow   = (const int*)d_in[2];
    const int*   ecol   = (const int*)d_in[3];
    const float* eattr  = (const float*)d_in[4];
    const float* emb_w  = (const float*)d_in[5];
    const float* emb_b  = (const float*)d_in[6];
    const float* temb   = (const float*)d_in[7];
    const float* theta  = (const float*)d_in[8];
    const float* wu_w   = (const float*)d_in[9];
    const float* wu_b   = (const float*)d_in[10];
    const float* We1    = (const float*)d_in[11];
    const float* be1    = (const float*)d_in[12];
    const float* We2    = (const float*)d_in[13];
    const float* be2    = (const float*)d_in[14];
    const float* Wc1    = (const float*)d_in[15];
    const float* bc1    = (const float*)d_in[16];
    const float* Wc2    = (const float*)d_in[17];
    const float* Wv1    = (const float*)d_in[18];
    const float* bv1    = (const float*)d_in[19];
    const float* Wv2    = (const float*)d_in[20];
    const float* Wn1    = (const float*)d_in[21];
    const float* bn1    = (const float*)d_in[22];
    const float* Wn2    = (const float*)d_in[23];
    const float* bn2    = (const float*)d_in[24];
    float* out = (float*)d_out;

    // carve workspace
    size_t off = 0;
    char* wsb = (char*)d_ws;
    auto carve = [&](size_t bytes) -> void* {
        void* p = wsb + off; off += (bytes + 255) & ~(size_t)255; return p;
    };
    float*    hh     = (float*)carve((size_t)NNODE * NPT * 64 * 4);
    float*    hcur   = (float*)carve((size_t)NNODE * PPART * 64 * 4);
    float*    coord  = (float*)carve((size_t)NNODE * PPART * 3 * 4);
    float*    coord2 = (float*)carve((size_t)NNODE * PPART * 3 * 4);
    float*    xp     = (float*)carve((size_t)NNODE * NPT * 3 * 4);
    float*    lw     = (float*)carve((size_t)NNODE * PPART * 4);
    float*    lw2    = (float*)carve((size_t)NNODE * PPART * 4);
    float*    cagg   = (float*)carve((size_t)NNODE * PPART * 3 * 4);
    float*    vagg   = (float*)carve((size_t)NNODE * PPART * 3 * 4);
    float*    nagg   = (float*)carve((size_t)NNODE * PPART * 64 * 4);
    float*    cnt    = (float*)carve((size_t)NNODE * 4);
    _Float16* We1h   = (_Float16*)carve((size_t)NLAY * K1PAD * 64 * 2);
    _Float16* We2h   = (_Float16*)carve((size_t)NLAY * 64 * 64 * 2);
    _Float16* Wc1h   = (_Float16*)carve((size_t)NLAY * 64 * 64 * 2);
    _Float16* Wv1h   = (_Float16*)carve((size_t)NLAY * 64 * 64 * 2);
    _Float16* Wn1h   = (_Float16*)carve((size_t)NLAY * KNPAD * 64 * 2);
    _Float16* Wn2h   = (_Float16*)carve((size_t)NLAY * 64 * 64 * 2);

    auto grid1 = [](int n) { return (n + 255) / 256; };

    // --- weight prep: fp32 -> blocked/swizzled f16 (one v16h per lane fragment) ---
    for (int l = 0; l < NLAY; ++l) {
        convert_swizzle_kernel<<<grid1(K1PAD * 64), 256, 0, stream>>>(
            We1 + (size_t)l * 131 * 64, We1h + (size_t)l * K1PAD * 64, 131, K1PAD);
        convert_swizzle_kernel<<<grid1(64 * 64), 256, 0, stream>>>(
            We2 + (size_t)l * 64 * 64, We2h + (size_t)l * 64 * 64, 64, 64);
        convert_swizzle_kernel<<<grid1(64 * 64), 256, 0, stream>>>(
            Wc1 + (size_t)l * 64 * 64, Wc1h + (size_t)l * 64 * 64, 64, 64);
        convert_swizzle_kernel<<<grid1(64 * 64), 256, 0, stream>>>(
            Wv1 + (size_t)l * 64 * 64, Wv1h + (size_t)l * 64 * 64, 64, 64);
        convert_swizzle_kernel<<<grid1(KNPAD * 64), 256, 0, stream>>>(
            Wn1 + (size_t)l * 128 * 64, Wn1h + (size_t)l * KNPAD * 64, 128, KNPAD);
        convert_swizzle_kernel<<<grid1(64 * 64), 256, 0, stream>>>(
            Wn2 + (size_t)l * 64 * 64, Wn2h + (size_t)l * 64 * 64, 64, 64);
    }

    // --- init ---
    zero_kernel<<<grid1(NNODE * PPART * 3), 256, 0, stream>>>(cagg, NNODE * PPART * 3);
    zero_kernel<<<grid1(NNODE * PPART * 3), 256, 0, stream>>>(vagg, NNODE * PPART * 3);
    zero_kernel<<<grid1(NNODE * PPART * 64), 256, 0, stream>>>(nagg, NNODE * PPART * 64);
    zero_kernel<<<grid1(NNODE), 256, 0, stream>>>(cnt, NNODE);
    count_kernel<<<grid1(NEDGE), 256, 0, stream>>>(erow, cnt);
    clip_cnt_kernel<<<grid1(NNODE), 256, 0, stream>>>(cnt);
    init_xp_kernel<<<grid1(NNODE * NPT * 3), 256, 0, stream>>>(x_in, xp);
    init_coord_lw_kernel<<<grid1(NNODE * PPART), 256, 0, stream>>>(x_in, coord, lw);
    embed_kernel<<<grid1(NNODE * NPT * 64), 256, 0, stream>>>(h_in, emb_w, emb_b, temb, hh);

    const int gridE = (NEDGE * PPART + 63) / 64;   // 16 rows/wave, 4 waves/block
    const int gridN = (NNODE * PPART + 63) / 64;

    // --- main particle-filter loop ---
    for (int k = 0; k < NPT; ++k) {
        bcast_kernel<<<grid1(NNODE * PPART * 64), 256, 0, stream>>>(hcur, hh, k);
        for (int i = 0; i < NLAY; ++i) {
            edge_gcl_kernel<<<gridE, 128, 0, stream>>>(
                hcur, coord, erow, ecol, eattr,
                We1h + (size_t)i * K1PAD * 64, be1 + i * 64,
                We2h + (size_t)i * 64 * 64,   be2 + i * 64,
                Wc1h + (size_t)i * 64 * 64,   bc1 + i * 64, Wc2 + i * 64,
                Wv1h + (size_t)i * 64 * 64,   bv1 + i * 64, Wv2 + i * 64,
                cagg, vagg, nagg);
            node_gcl_kernel<<<gridN, 128, 0, stream>>>(
                hcur, coord, cnt, cagg, vagg, nagg,
                Wn1h + (size_t)i * KNPAD * 64, bn1 + i * 64,
                Wn2h + (size_t)i * 64 * 64,    bn2 + i * 64,
                k * NLAY + i);
        }
        if (k != NPT - 1) {
            weight_kernel<<<grid1(NNODE), 256, 0, stream>>>(wu_w, wu_b, coord, xp, lw, lw2, k);
            sample_kernel<<<grid1(NNODE), 256, 0, stream>>>(coord, coord2, lw2, lw, xp, k);
            copy_kernel<<<grid1(NNODE * PPART * 3), 256, 0, stream>>>(coord, coord2,
                                                                      NNODE * PPART * 3);
        }
    }

    final_kernel<<<grid1(NNODE * 3), 256, 0, stream>>>(theta, xp, out);
}
